// RNN_20847771255413
// MI455X (gfx1250) — matrix operand order repaired and checked by reading, not compiled, not run on hardware
//
#include <hip/hip_runtime.h>
#include <hip/hip_bf16.h>

typedef __attribute__((ext_vector_type(16))) _Float16 v16h;
typedef __attribute__((ext_vector_type(8)))  float    v8f;

#define RNN_N   27
#define RNN_T   1024
#define RNN_B   2048
#define KP      32
#define NTILES  128   // RNN_B / 16
#define PFDIST  8     // xp prefetch distance (timesteps)

// ---- activations: CDNA5 hardware V_TANH_F32 (TRANS pipe, co-executes with
// XDL WMMA). Sigmoid gates arrive pre-scaled by 0.5 from the GEMM, so
// sigmoid(z) = fma(0.5, tanh(z_prescaled), 0.5) -- no extra multiply.
#if __has_builtin(__builtin_amdgcn_tanhf)
__device__ __forceinline__ float tanh_f(float x) {
    return __builtin_amdgcn_tanhf(x);
}
#else
__device__ __forceinline__ float tanh_f(float x) {
    const float e = __expf(-2.0f * x);
    return __builtin_fmaf(2.0f, __builtin_amdgcn_rcpf(1.0f + e), -1.0f);
}
#endif
__device__ __forceinline__ float sigm_pre(float zhalf) {   // zhalf == 0.5*z
    return __builtin_fmaf(0.5f, tanh_f(zhalf), 0.5f);
}

// A-fragment (16x32 f16) K index for vector element e, lane half hi (lane>=16)
__device__ __forceinline__ int a_k(int e, int hi) {
    return e + ((e >> 3) << 3) + (hi << 3);
}
// B-fragment (32x16 f16) K index: same order as A fragment
__device__ __forceinline__ int b_k(int e, int hi) {
    return e + ((e >> 3) << 3) + (hi << 3);
}

__device__ __forceinline__ v8f wmma_f16(v16h a, v16h b, v8f c) {
    return __builtin_amdgcn_wmma_f32_16x16x32_f16(false, a, false, b,
                                                  (short)0, c, false, false);
}

// ---------------------------------------------------------------------------
// Prepack x [B,T,27] f32 -> xp in per-lane A-fragment order (f16): the scan
// loop then reads one coalesced 32B v16h per lane per timestep.
// gid = (tile*T + t)*32 + lane
// ---------------------------------------------------------------------------
__global__ __launch_bounds__(256)
void rnn_xpack_kernel(const float* __restrict__ x, _Float16* __restrict__ xp) {
    const int gid  = blockIdx.x * 256 + threadIdx.x;
    const int lane = gid & 31;
    const int row  = lane & 15;
    const int hi   = lane >> 4;
    const int tt   = gid >> 5;              // tile*T + t
    const int t    = tt & (RNN_T - 1);
    const int tile = tt >> 10;
    const float* src = x + ((size_t)(tile * 16 + row) * RNN_T + t) * RNN_N;
    v16h v;
    #pragma unroll
    for (int e = 0; e < 16; ++e) {
        const int k = a_k(e, hi);
        const float f = src[k < RNN_N ? k : 0];     // clamped load, no divergence
        v[e] = (_Float16)(k < RNN_N ? f : 0.0f);
    }
    *(v16h*)(xp + (size_t)gid * 16) = v;
}

// ---------------------------------------------------------------------------
// Fused LSTM scan + i2o projection. One wave32 per 16 batch rows; weights live
// in VGPRs as WMMA B-fragments (sigmoid slabs pre-scaled by 0.5); h transposed
// D->A layout via a same-wave LDS bounce; x fragment double-buffered +
// global_prefetch_b8 look-ahead.
// ---------------------------------------------------------------------------
__global__ __launch_bounds__(32)
void rnn_lstm_wmma_kernel(const float* __restrict__ x,
                          const _Float16* __restrict__ xp, int use_xp,
                          const float* __restrict__ h0,
                          const float* __restrict__ c0,
                          const float* __restrict__ Wih,
                          const float* __restrict__ Whh,
                          const float* __restrict__ bih,
                          const float* __restrict__ bhh,
                          const float* __restrict__ Wout,
                          const float* __restrict__ bout,
                          float* __restrict__ out)
{
    __shared__ _Float16 hbuf[16][KP];

    const int lane = threadIdx.x;
    const int row  = lane & 15;
    const int hi   = lane >> 4;
    const int tile = blockIdx.x;
    const int b0   = tile * 16;

    // ---- time-invariant weight B-fragments, kept in VGPRs ----
    // z layout: 4 gate slabs (i,f,g,o) x 32 padded cols -> 8 N-tiles, j=2*gate+s
    // Sigmoid gates (i,f,o) pre-scaled by 0.5 (exact in fp16: exponent shift).
    v16h Bih[8], Bhh[8], Bo[2];
    float zbias[8], obias[2];
    #pragma unroll
    for (int j = 0; j < 8; ++j) {
        const int gate = j >> 1;
        const float gs = (gate == 2) ? 1.0f : 0.5f;  // g-gate unscaled
        const int nl   = ((j & 1) << 4) + row;       // col within 32-wide slab
        const int wrow = gate * RNN_N + nl;          // W row (valid iff nl<27)
        v16h bi, bh;
        #pragma unroll
        for (int e = 0; e < 16; ++e) {
            const int  k  = b_k(e, hi);
            const bool ok = (nl < RNN_N) && (k < RNN_N);
            const int  idx = ok ? (wrow * RNN_N + k) : 0;
            const float wi = Wih[idx], wh = Whh[idx];
            bi[e] = (_Float16)(ok ? gs * wi : 0.0f);
            bh[e] = (_Float16)(ok ? gs * wh : 0.0f);
        }
        Bih[j] = bi; Bhh[j] = bh;
        zbias[j] = (nl < RNN_N) ? gs * (bih[wrow] + bhh[wrow]) : 0.0f;
    }
    #pragma unroll
    for (int j = 0; j < 2; ++j) {
        const int m = (j << 4) + row;                // output col (valid iff <27)
        v16h bo;
        #pragma unroll
        for (int e = 0; e < 16; ++e) {
            const int  k  = b_k(e, hi);
            const bool ok = (m < RNN_N) && (k < RNN_N);
            const float w = Wout[ok ? (m * RNN_N + k) : 0];
            bo[e] = (_Float16)(ok ? w : 0.0f);
        }
        Bo[j] = bo;
        obias[j] = (m < RNN_N) ? bout[m] : 0.0f;
    }

    // ---- initial state: h0 as A-fragment, c0 as two C/D fragments ----
    v16h Ah;
    #pragma unroll
    for (int e = 0; e < 16; ++e) {
        const int k = a_k(e, hi);
        const float f = h0[(size_t)(b0 + row) * RNN_N + (k < RNN_N ? k : 0)];
        Ah[e] = (_Float16)(k < RNN_N ? f : 0.0f);
    }
    v8f cfr[2];
    #pragma unroll
    for (int s = 0; s < 2; ++s) {
        const int n = (s << 4) + row;
        #pragma unroll
        for (int r = 0; r < 8; ++r) {
            const int b = b0 + r + (hi << 3);
            const float f = c0[(size_t)b * RNN_N + (n < RNN_N ? n : 0)];
            cfr[s][r] = (n < RNN_N) ? f : 0.0f;
        }
    }
    v8f hD[2];
    hD[0] = (v8f){}; hD[1] = (v8f){};

    float* hn = out + (size_t)RNN_B * RNN_T * RNN_N;
    float* cn = hn + (size_t)RNN_B * RNN_N;

    // ---- x fragment loader (packed fast path / direct fallback) ----
    auto load_ax = [&](int t) -> v16h {
        if (use_xp) {
            return *(const v16h*)(xp + ((size_t)(tile * RNN_T + t) * 32 + lane) * 16);
        }
        const float* src = x + ((size_t)(b0 + row) * RNN_T + t) * RNN_N;
        v16h a;
        #pragma unroll
        for (int e = 0; e < 16; ++e) {
            const int k = a_k(e, hi);
            const float f = src[k < RNN_N ? k : 0];
            a[e] = (_Float16)(k < RNN_N ? f : 0.0f);
        }
        return a;
    };

    v16h Ax = load_ax(0);                 // prologue of the software pipeline

    #pragma clang loop unroll(disable)
    for (int t = 0; t < RNN_T; ++t) {
        // issue next timestep's load now; a full step of WMMA+TRANS hides it
        const int tn = (t + 1 < RNN_T) ? (t + 1) : (RNN_T - 1);
        v16h Axn = load_ax(tn);
        if (use_xp) {
            // deep look-ahead into L0/L2: global_prefetch_b8, no counter cost
            const int tp = (t + PFDIST < RNN_T) ? (t + PFDIST) : (RNN_T - 1);
            __builtin_prefetch(xp + ((size_t)(tile * RNN_T + tp) * 32 + lane) * 16,
                               0, 0);
        }

        // ---- z = x W_ih^T + h W_hh^T + b ; gates; state update ----
        #pragma unroll
        for (int s = 0; s < 2; ++s) {
            v8f zi, zf, zg, zo;
            #pragma unroll
            for (int r = 0; r < 8; ++r) {
                zi[r] = zbias[0 + s]; zf[r] = zbias[2 + s];
                zg[r] = zbias[4 + s]; zo[r] = zbias[6 + s];
            }
            zi = wmma_f16(Ax, Bih[0 + s], zi); zi = wmma_f16(Ah, Bhh[0 + s], zi);
            zf = wmma_f16(Ax, Bih[2 + s], zf); zf = wmma_f16(Ah, Bhh[2 + s], zf);
            zg = wmma_f16(Ax, Bih[4 + s], zg); zg = wmma_f16(Ah, Bhh[4 + s], zg);
            zo = wmma_f16(Ax, Bih[6 + s], zo); zo = wmma_f16(Ah, Bhh[6 + s], zo);
            #pragma unroll
            for (int r = 0; r < 8; ++r) {
                const float ii = sigm_pre(zi[r]);        // z pre-scaled by GEMM
                const float ff = sigm_pre(zf[r]);
                const float gg = tanh_f(zg[r]);
                const float oo = sigm_pre(zo[r]);
                const float cc = __builtin_fmaf(ff, cfr[s][r], ii * gg);
                cfr[s][r] = cc;
                hD[s][r]  = oo * tanh_f(cc);
            }
        }

        // ---- transpose h: D-layout -> LDS -> A-layout (same-wave, in-order DS)
        #pragma unroll
        for (int s = 0; s < 2; ++s)
            #pragma unroll
            for (int r = 0; r < 8; ++r)
                hbuf[r + (hi << 3)][(s << 4) + row] = (_Float16)hD[s][r];
        __syncthreads();   // 1-wave block: lowers to s_wait_dscnt + NOP barrier
        #pragma unroll
        for (int e = 0; e < 16; ++e)
            Ah[e] = hbuf[row][a_k(e, hi)];

        // ---- fused i2o projection: out_t = h_new @ W_out^T + b_out ----
        #pragma unroll
        for (int j = 0; j < 2; ++j) {
            v8f oacc;
            #pragma unroll
            for (int r = 0; r < 8; ++r) oacc[r] = obias[j];
            oacc = wmma_f16(Ah, Bo[j], oacc);
            const int col = (j << 4) + row;
            if (col < RNN_N) {
                #pragma unroll
                for (int r = 0; r < 8; ++r) {
                    const int b = b0 + r + (hi << 3);
                    out[((size_t)b * RNN_T + t) * RNN_N + col] = oacc[r];
                }
            }
        }

        Ax = Axn;
    }

    // ---- final hn, cn ----
    #pragma unroll
    for (int s = 0; s < 2; ++s) {
        const int n = (s << 4) + row;
        if (n < RNN_N) {
            #pragma unroll
            for (int r = 0; r < 8; ++r) {
                const int b = b0 + r + (hi << 3);
                hn[(size_t)b * RNN_N + n] = hD[s][r];
                cn[(size_t)b * RNN_N + n] = cfr[s][r];
            }
        }
    }
}

extern "C" void kernel_launch(void* const* d_in, const int* in_sizes, int n_in,
                              void* d_out, int out_size, void* d_ws, size_t ws_size,
                              hipStream_t stream) {
    const float* x    = (const float*)d_in[0];
    const float* h0   = (const float*)d_in[1];
    const float* c0   = (const float*)d_in[2];
    const float* Wih  = (const float*)d_in[3];
    const float* Whh  = (const float*)d_in[4];
    const float* bih  = (const float*)d_in[5];
    const float* bhh  = (const float*)d_in[6];
    const float* Wout = (const float*)d_in[7];
    const float* bout = (const float*)d_in[8];
    float* out = (float*)d_out;

    const size_t xp_bytes = (size_t)NTILES * RNN_T * 32 * 16 * sizeof(_Float16);
    const int use_xp = (ws_size >= xp_bytes) ? 1 : 0;
    _Float16* xp = (_Float16*)d_ws;

    if (use_xp) {
        rnn_xpack_kernel<<<NTILES * RNN_T * 32 / 256, 256, 0, stream>>>(x, xp);
    }
    rnn_lstm_wmma_kernel<<<NTILES, 32, 0, stream>>>(x, xp, use_xp, h0, c0,
                                                    Wih, Whh, bih, bhh,
                                                    Wout, bout, out);
}
